// CRF_76940044141098
// MI455X (gfx1250) — matrix operand order repaired and checked by reading, not compiled, not run on hardware
//
#include <hip/hip_runtime.h>
#include <hip/hip_bf16.h>
#include <stdint.h>

typedef __attribute__((ext_vector_type(16))) _Float16    v16h;
typedef __attribute__((ext_vector_type(8)))  float       v8f;
typedef __attribute__((ext_vector_type(4)))  unsigned int v4u;
typedef __attribute__((ext_vector_type(8)))  int         v8i;
typedef __attribute__((ext_vector_type(4)))  int         v4i;

#define NEG_INF (-10000.0f)
#define LDIM  128          // label count (reference: 128)
#define BTILE 16           // batch rows per workgroup (one WMMA M-tile)
#define NWAVE 4            // waves per block (wave32 -> 128 threads)
#define TPW   2            // output i-tiles per wave (8 tiles / 4 waves)
#define LOG2E 1.4426950408889634f
#define LN2   0.6931471805599453f

#if __has_builtin(__builtin_amdgcn_tensor_load_to_lds)
#define USE_TDM 1
#else
#define USE_TDM 0
#endif

#if USE_TDM
// Issue one TDM DMA: 16 rows x 128 f32, row stride S*LDIM elements -> LDS.
// D# built per CDNA5 ISA 8.3/8.4 (group0: count/lds/global/type, group1: dims).
__device__ __forceinline__ void tdm_load_tile(const float* gsrc,
                                              uint32_t lds_off, int S)
{
    const uint64_t ga = (uint64_t)(uintptr_t)gsrc;
    v4u g0;
    g0.x = 1u;                                      // count=1 (valid user D#)
    g0.y = lds_off;                                 // lds_addr (bytes)
    g0.z = (uint32_t)(ga & 0xffffffffu);            // global_addr[31:0]
    g0.w = (uint32_t)((ga >> 32) & 0x01ffffffu)     // global_addr[56:32]
         | (2u << 30);                              // type=2 ("image")
    v8i g1;
    g1[0] = (int)(2u << 16);                        // data_size=4B, mask=0
    g1[1] = (int)((uint32_t)LDIM  << 16);           // tensor_dim0 = 128
    g1[2] = (int)((uint32_t)BTILE << 16);           // tensor_dim1 = 16
    g1[3] = (int)((uint32_t)LDIM  << 16);           // tile_dim0   = 128
    g1[4] = (int)(uint32_t)BTILE;                   // tile_dim1   = 16
    g1[5] = (int)(uint32_t)(S * LDIM);              // dim0_stride lo32
    g1[6] = 0;                                      // stride hi bits = 0
    g1[7] = 0;
    v4i gz = {0, 0, 0, 0};
#if __clang_major__ >= 23
    v8i gz8 = {0, 0, 0, 0, 0, 0, 0, 0};
    __builtin_amdgcn_tensor_load_to_lds(g0, g1, gz, gz, gz8, 0);
#else
    __builtin_amdgcn_tensor_load_to_lds(g0, g1, gz, gz, 0);
#endif
}
#endif

__global__ __launch_bounds__(128) void crf_fwd_wmma_kernel(
    const float* __restrict__ logits,       // [B, S, L]
    const float* __restrict__ transitions,  // [L, L]
    const int*   __restrict__ lens,         // [B]
    float*       __restrict__ out,          // [B]
    int B, int S)
{
    __shared__ float    alpha[BTILE][LDIM];        // LDS mirror of alpha (f32)
    __shared__ _Float16 vbuf[BTILE][LDIM];         // exp2-scaled alpha, f16
    __shared__ float    logit_lds[2][BTILE][LDIM]; // TDM double buffer
    __shared__ float    pred[BTILE][8];            // partial reductions
    __shared__ float    rowmax[BTILE];
    __shared__ int      lens_s[BTILE];

    const int tid  = threadIdx.x;
    const int lane = tid & 31;
    const int wave = tid >> 5;
    const int gb0  = blockIdx.x * BTILE;

    // D/C-matrix ownership of this lane (16x16 f32 layout):
    //   VGPR r -> row M = r (lanes 0-15) or 8+r (lanes 16-31); col N = lane&15
    const int brow0 = (lane < 16) ? 0 : 8;
    int icol[TPW];
#pragma unroll
    for (int tt = 0; tt < TPW; ++tt)
        icol[tt] = (wave * TPW + tt) * 16 + (lane & 15);

    // ---- init: lane-resident alpha values + LDS mirror ----
    float aval[TPW][8];
#pragma unroll
    for (int tt = 0; tt < TPW; ++tt)
#pragma unroll
        for (int r = 0; r < 8; ++r) {
            aval[tt][r] = (icol[tt] == 0) ? 0.0f : NEG_INF;  // BOS_IDX = 0
            alpha[brow0 + r][icol[tt]] = aval[tt][r];
        }
    if (tid < BTILE) {
        int gb = gb0 + tid;
        lens_s[tid] = (gb < B) ? lens[gb] : 0;
    }
    __syncthreads();

    int len_r[8];
#pragma unroll
    for (int r = 0; r < 8; ++r) len_r[r] = lens_s[brow0 + r];

    const float* logit_base = logits + (size_t)gb0 * S * LDIM;

#if USE_TDM
    // Prime the pipeline: DMA step-0 logit tile into buffer 0 (wave 0 only;
    // TDM ignores EXEC, one DMA per wave instruction).
    if (wave == 0)
        tdm_load_tile(logit_base, (uint32_t)(uintptr_t)&logit_lds[0][0][0], S);
#endif

    // ---- precompute B-matrix fragments: E^T[j,i] = exp(T[i,j]) ----
    // B (KxN=32x16) f16 layout: halves 0..7 -> K = (lane<16?0:8)+h,
    // halves 8..15 -> K = 16+(lane<16?0:8)+(h-8); column N = lane & 15.
    v16h bfrag[TPW][4];
#pragma unroll
    for (int tt = 0; tt < TPW; ++tt) {
#pragma unroll
        for (int c = 0; c < 4; ++c) {
            v16h f;
#pragma unroll
            for (int h = 0; h < 16; ++h) {
                int j = c * 32 + ((lane < 16) ? 0 : 8) + h + ((h < 8) ? 0 : 8);
                f[h] = (_Float16)__builtin_amdgcn_exp2f(
                           transitions[icol[tt] * LDIM + j] * LOG2E);
            }
            bfrag[tt][c] = f;
        }
    }

    // ---- sequential scan over S ----
    for (int t = 0; t < S; ++t) {
        __syncthreads();  // alpha stable; buf[(t+1)&1] consumers (t-1) done

#if USE_TDM
        if (wave == 0 && t + 1 < S)   // DMA next step's tile into other buffer
            tdm_load_tile(logit_base + (size_t)(t + 1) * LDIM,
                          (uint32_t)(uintptr_t)&logit_lds[(t + 1) & 1][0][0], S);
#endif

        // phase A: per-row max over 128 labels (8 threads x 16 elems per row)
        {
            int row = tid >> 3, seg = tid & 7;
            float m = -3.0e38f;
#pragma unroll
            for (int k = 0; k < 16; ++k)
                m = fmaxf(m, alpha[row][seg * 16 + k]);
            pred[row][seg] = m;
        }
        __syncthreads();
        if (tid < BTILE) {
            float m = pred[tid][0];
#pragma unroll
            for (int s = 1; s < 8; ++s) m = fmaxf(m, pred[tid][s]);
            rowmax[tid] = m;
        }
        __syncthreads();

        // phase B: v = exp(alpha - m) -> f16 (raw v_exp_f32, inputs <= 0)
        for (int idx = tid; idx < BTILE * LDIM; idx += blockDim.x) {
            int b = idx >> 7, j = idx & (LDIM - 1);
            vbuf[b][j] = (_Float16)__builtin_amdgcn_exp2f(
                             (alpha[b][j] - rowmax[b]) * LOG2E);
        }

#if USE_TDM
        if (wave == 0) {              // ensure buf[t&1] landed before release
            if (t + 1 < S) __builtin_amdgcn_s_wait_tensorcnt(1);
            else           __builtin_amdgcn_s_wait_tensorcnt(0);
        }
#endif
        __syncthreads();              // publishes vbuf + logit buffer

        // Hoist this lane's rowmax + logit values for its 16 D elements.
        float rm[8];
#pragma unroll
        for (int r = 0; r < 8; ++r) rm[r] = rowmax[brow0 + r];

        float lgv[TPW][8];
#pragma unroll
        for (int tt = 0; tt < TPW; ++tt)
#pragma unroll
            for (int r = 0; r < 8; ++r) {
#if USE_TDM
                lgv[tt][r] = logit_lds[t & 1][brow0 + r][icol[tt]];
#else
                lgv[tt][r] =
                    logit_base[((size_t)(brow0 + r) * S + t) * LDIM + icol[tt]];
#endif
            }

        // phase C: A fragments (16x32 f16 per k-chunk), shared by both tiles.
        // A layout: row M = lane&15; halves 0..7 -> K = base+(lane<16?0:8)+h,
        //           halves 8..15 -> K = base+16+(lane<16?0:8)+(h-8).
        v16h afrag[4];
#pragma unroll
        for (int c = 0; c < 4; ++c) {
            const int row = lane & 15;
            const int ko  = (lane < 16) ? 0 : 8;
            v16h a;
#pragma unroll
            for (int h = 0; h < 8; ++h)
                a[h] = vbuf[row][c * 32 + ko + h];
#pragma unroll
            for (int h = 8; h < 16; ++h)
                a[h] = vbuf[row][c * 32 + 16 + ko + (h - 8)];
            afrag[c] = a;
        }

        // All 8 WMMAs back-to-back: tile 1's chain provides the required
        // co-exec distance before tile 0's accumulator is consumed.
        v8f acc[TPW];
#pragma unroll
        for (int tt = 0; tt < TPW; ++tt) {
            v8f a = {};
#pragma unroll
            for (int c = 0; c < 4; ++c) {
                a = __builtin_amdgcn_wmma_f32_16x16x32_f16(
                    /*neg_a=*/false, afrag[c],
                    /*neg_b=*/false, bfrag[tt][c],
                    /*c_mod=*/(short)0, a,
                    /*reuse_a=*/false, /*reuse_b=*/false);
            }
            acc[tt] = a;
        }

        // Branchless masked update: aval += m * (anew - aval), m in {0,1}.
        // The fma keeps anew an unconditional operand -> no exec branching.
#pragma unroll
        for (int tt = 0; tt < TPW; ++tt) {
#pragma unroll
            for (int r = 0; r < 8; ++r) {
                // raw v_log_f32: acc > 0 and normal by construction
                const float anew = lgv[tt][r] + rm[r]
                                 + LN2 * __builtin_amdgcn_logf(acc[tt][r]);
                const float m = (t < len_r[r]) ? 1.0f : 0.0f;
                aval[tt][r] = fmaf(m, anew - aval[tt][r], aval[tt][r]);
                alpha[brow0 + r][icol[tt]] = aval[tt][r];
            }
        }
    }

    // ---- final logsumexp over labels per batch row ----
    __syncthreads();
    {
        int row = tid >> 3, seg = tid & 7;
        float m = -3.0e38f;
#pragma unroll
        for (int k = 0; k < 16; ++k)
            m = fmaxf(m, alpha[row][seg * 16 + k]);
        pred[row][seg] = m;
    }
    __syncthreads();
    if (tid < BTILE) {
        float m = pred[tid][0];
#pragma unroll
        for (int s = 1; s < 8; ++s) m = fmaxf(m, pred[tid][s]);
        rowmax[tid] = m;
    }
    __syncthreads();
    {
        int row = tid >> 3, seg = tid & 7;
        float p = 0.0f;
#pragma unroll
        for (int k = 0; k < 16; ++k)
            p += __builtin_amdgcn_exp2f(
                     (alpha[row][seg * 16 + k] - rowmax[row]) * LOG2E);
        pred[row][seg] = p;
    }
    __syncthreads();
    if (tid < BTILE) {
        int gb = gb0 + tid;
        if (gb < B) {
            float s = 0.0f;
#pragma unroll
            for (int k = 0; k < 8; ++k) s += pred[tid][k];
            out[gb] = rowmax[tid] + LN2 * __builtin_amdgcn_logf(s);
        }
    }
}

extern "C" void kernel_launch(void* const* d_in, const int* in_sizes, int n_in,
                              void* d_out, int out_size, void* d_ws, size_t ws_size,
                              hipStream_t stream) {
    const float* logits      = (const float*)d_in[0];
    const float* transitions = (const float*)d_in[1];
    const int*   lens        = (const int*)d_in[2];
    float*       out         = (float*)d_out;

    const int B = in_sizes[2];                 // lens: [B]
    const int S = in_sizes[0] / (B * LDIM);    // logits: [B, S, 128]

    const int blocks = (B + BTILE - 1) / BTILE;
    crf_fwd_wmma_kernel<<<blocks, NWAVE * 32, 0, stream>>>(
        logits, transitions, lens, out, B, S);
}
